// MaxEntMCMC_15195594293809
// MI455X (gfx1250) — matrix operand order repaired and checked by reading, not compiled, not run on hardware
//
#include <hip/hip_runtime.h>
#include <hip/hip_bf16.h>
#include <stdint.h>

typedef __attribute__((ext_vector_type(16))) _Float16 v16h;
typedef __attribute__((ext_vector_type(8)))  _Float16 v8h;
typedef __attribute__((ext_vector_type(8)))  float    v8f;

#define BM 128
#define BN 128
#define BK 32
#define LDS_STRIDE 40   // 32 f16 + 8 f16 pad per row (80B rows, 16B aligned)

// ---------------------------------------------------------------------------
// counter-based hash RNG -> uniform [0,1)
// ---------------------------------------------------------------------------
__device__ __forceinline__ float rng_uniform(uint32_t seed, uint32_t idx) {
    uint32_t x = idx ^ (seed * 0x9E3779B9u);
    x *= 0x85EBCA6Bu;  x ^= x >> 13;
    x *= 0xC2B2AE35u;  x ^= x >> 16;
    x += seed * 0x27D4EB2Fu;
    x ^= x >> 15;      x *= 0x2C1B3C6Du;
    x ^= x >> 12;
    return (float)(x >> 8) * (1.0f / 16777216.0f);
}

// ---------------------------------------------------------------------------
// Js = 0.5*(J + J^T), zero diagonal, f32 -> f16   (Js is symmetric!)
// ---------------------------------------------------------------------------
__global__ void sym_kernel(const float* __restrict__ J,
                           _Float16* __restrict__ Js, int N) {
    int idx = blockIdx.x * blockDim.x + threadIdx.x;
    if (idx >= N * N) return;
    int i = idx / N, j = idx % N;
    float v = (i == j) ? 0.0f : 0.5f * (J[i * N + j] + J[j * N + i]);
    Js[idx] = (_Float16)v;
}

// s0 = (init_u > 0.5) as f16
__global__ void init_kernel(const float* __restrict__ u,
                            _Float16* __restrict__ s, int total) {
    int idx = blockIdx.x * blockDim.x + threadIdx.x;
    if (idx < total) s[idx] = (u[idx] > 0.5f) ? (_Float16)1.0f : (_Float16)0.0f;
}

// ---------------------------------------------------------------------------
// One synchronous Gibbs sweep, fused:  C = S @ Js  (Js symmetric),
// probs = sigmoid(h + C), Snext = (rng < probs), optional f32 dump to Out.
// Block: 256 threads = 8 wave32s; 128x128 tile; wave = 32x64 (2x4 WMMA tiles).
// ---------------------------------------------------------------------------
__global__ __launch_bounds__(256)
void gibbs_sweep_kernel(const _Float16* __restrict__ S,    // (B,N) f16 state
                        const _Float16* __restrict__ Js,   // (N,N) f16 symmetric
                        const float*    __restrict__ h,    // (N)
                        _Float16*       __restrict__ Snext,// (B,N) f16
                        float*          __restrict__ Out,  // (B,N) f32 or null
                        int N, uint32_t seed)
{
    __shared__ __attribute__((aligned(16))) _Float16 As[BM * LDS_STRIDE];
    __shared__ __attribute__((aligned(16))) _Float16 Bs[BN * LDS_STRIDE];

    const int tid   = threadIdx.x;
    const int wave  = tid >> 5;
    const int lane  = tid & 31;
    const int laneM = lane & 15;
    const int half  = lane >> 4;

    const int bBase = blockIdx.x * BM;          // batch-row tile base
    const int nBase = blockIdx.y * BN;          // unit-column tile base
    const int waveM = (wave & 3) * 32;          // 4 waves along M
    const int waveN = (wave >> 2) * 64;         // 2 waves along N

    v8f acc[2][4];
    #pragma unroll
    for (int i = 0; i < 2; ++i)
        #pragma unroll
        for (int j = 0; j < 4; ++j)
            #pragma unroll
            for (int e = 0; e < 8; ++e) acc[i][j][e] = 0.0f;

    for (int k0 = 0; k0 < N; k0 += BK) {
        // --- stage A (state rows) and B (Js rows == Js cols, symmetric) ---
        #pragma unroll
        for (int it = 0; it < 2; ++it) {
            int idx = tid + it * 256;                  // 0..511
            int row = idx >> 2;                        // 0..127
            int ch  = idx & 3;                         // 8-f16 chunk
            v8h a = *(const v8h*)(S  + (size_t)(bBase + row) * N + k0 + ch * 8);
            *(v8h*)(As + row * LDS_STRIDE + ch * 8) = a;
            v8h b = *(const v8h*)(Js + (size_t)(nBase + row) * N + k0 + ch * 8);
            *(v8h*)(Bs + row * LDS_STRIDE + ch * 8) = b;
        }
        __syncthreads();

        // --- A fragments: lane row = laneM, K = half*8+0..7 and 16+half*8+0..7
        v16h afrag[2], bfrag[4];
        #pragma unroll
        for (int mt = 0; mt < 2; ++mt) {
            const _Float16* p = As + (waveM + mt * 16 + laneM) * LDS_STRIDE + half * 8;
            v8h lo = *(const v8h*)(p);
            v8h hi = *(const v8h*)(p + 16);
            #pragma unroll
            for (int e = 0; e < 8; ++e) { afrag[mt][e] = lo[e]; afrag[mt][8 + e] = hi[e]; }
        }
        // --- B fragments: lane col = laneM, K = half*8+0..7 and 16+half*8+0..7
        #pragma unroll
        for (int nt = 0; nt < 4; ++nt) {
            const _Float16* p = Bs + (waveN + nt * 16 + laneM) * LDS_STRIDE + half * 8;
            v8h lo = *(const v8h*)(p);
            v8h hi = *(const v8h*)(p + 16);
            #pragma unroll
            for (int e = 0; e < 8; ++e) { bfrag[nt][e] = lo[e]; bfrag[nt][8 + e] = hi[e]; }
        }

        #pragma unroll
        for (int mt = 0; mt < 2; ++mt)
            #pragma unroll
            for (int nt = 0; nt < 4; ++nt)
                acc[mt][nt] = __builtin_amdgcn_wmma_f32_16x16x32_f16(
                    false, afrag[mt], false, bfrag[nt],
                    (short)0, acc[mt][nt], false, false);
        __syncthreads();
    }

    // --- fused epilogue: sigmoid + Bernoulli sample ------------------------
    // C/D layout: n = laneM; m = vgpr + 8*half (within a 16x16 tile)
    #pragma unroll
    for (int nt = 0; nt < 4; ++nt) {
        int n = nBase + waveN + nt * 16 + laneM;
        float hn = h[n];
        #pragma unroll
        for (int mt = 0; mt < 2; ++mt) {
            #pragma unroll
            for (int r = 0; r < 8; ++r) {
                int m = bBase + waveM + mt * 16 + r + half * 8;
                float field = hn + acc[mt][nt][r];
                float p = 1.0f / (1.0f + __expf(-field));
                float u = rng_uniform(seed, (uint32_t)m * (uint32_t)N + (uint32_t)n);
                float sNew = (u < p) ? 1.0f : 0.0f;
                Snext[(size_t)m * N + n] = (_Float16)sNew;
                if (Out) Out[(size_t)m * N + n] = sNew;
            }
        }
    }
}

// ---------------------------------------------------------------------------
extern "C" void kernel_launch(void* const* d_in, const int* in_sizes, int n_in,
                              void* d_out, int out_size, void* d_ws, size_t ws_size,
                              hipStream_t stream) {
    const float* h  = (const float*)d_in[0];   // (N)
    const float* J  = (const float*)d_in[1];   // (N,N)
    const float* u0 = (const float*)d_in[2];   // (B,N)
    const int N = in_sizes[0];
    const int B = in_sizes[2] / N;
    const int BURNIN = 20;                     // reference constants
    const int STEPS  = 5;

    // workspace layout: Js f16 (N*N) | state0 f16 (B*N) | state1 f16 (B*N)
    _Float16* Js = (_Float16*)d_ws;
    _Float16* s0 = Js + (size_t)N * N;
    _Float16* s1 = s0 + (size_t)B * N;
    float* out = (float*)d_out;

    { int total = N * N;
      sym_kernel<<<(total + 255) / 256, 256, 0, stream>>>(J, Js, N); }
    { int total = B * N;
      init_kernel<<<(total + 255) / 256, 256, 0, stream>>>(u0, s0, total); }

    dim3 grid(B / BM, N / BN), block(256);
    _Float16* bufs[2] = { s0, s1 };
    for (int step = 0; step < BURNIN + STEPS; ++step) {
        _Float16* src = bufs[step & 1];
        _Float16* dst = bufs[(step + 1) & 1];
        float* o = (step >= BURNIN) ? out + (size_t)(step - BURNIN) * B * N : nullptr;
        uint32_t seed = 0x1234567u + (uint32_t)step * 0x9E3779B9u;
        gibbs_sweep_kernel<<<grid, block, 0, stream>>>(src, Js, h, dst, o, N, seed);
    }
}
